// RCNModel_55963423867007
// MI455X (gfx1250) — matrix operand order repaired and checked by reading, not compiled, not run on hardware
//
#include <hip/hip_runtime.h>
#include <hip/hip_bf16.h>
#include <math.h>

// ---------------- problem constants (match reference) ----------------
#define NNODES 65536
#define NEDGES 1048576
#define NGRAPH 1024
#define NPG    64

typedef __attribute__((ext_vector_type(16))) _Float16 v16h;
typedef __attribute__((ext_vector_type(8)))  _Float16 v8h;
typedef __attribute__((ext_vector_type(8)))  float    v8f;

// ---------------- small utility kernels ----------------
__global__ void fill_kernel(float* p, float v, int n) {
  int i = blockIdx.x * blockDim.x + threadIdx.x;
  if (i < n) p[i] = v;
}

// Weight prep: fp32 W[K][N] -> f16 transposed padded WT[Npad][Kpad] (zero pad)
__global__ void prep_wt_kernel(const float* __restrict__ W, _Float16* __restrict__ WT,
                               int K, int N, int Kpad, int Npad) {
  int idx = blockIdx.x * blockDim.x + threadIdx.x;
  if (idx >= Npad * Kpad) return;
  int n = idx / Kpad, k = idx % Kpad;
  float v = (k < K && n < N) ? W[(size_t)k * N + n] : 0.f;
  WT[idx] = (_Float16)v;
}

// h0 (f16, padded to 32): [emb[piece] | x[:,1:13] | 0,0,0,0]
__global__ void embed_kernel(const float* __restrict__ x,
                             const float* __restrict__ emb,
                             _Float16* __restrict__ h) {
  int i = blockIdx.x * blockDim.x + threadIdx.x;
  if (i >= NNODES) return;
  int piece = (int)x[i * 13];
  piece = piece < 0 ? 0 : (piece > 12 ? 12 : piece);
  for (int j = 0; j < 16; ++j) h[i * 32 + j] = (_Float16)emb[piece * 16 + j];
  for (int j = 0; j < 12; ++j) h[i * 32 + 16 + j] = (_Float16)x[i * 13 + 1 + j];
  for (int j = 28; j < 32; ++j) h[i * 32 + j] = (_Float16)0.f;
}

// ---------------- WMMA f16 GEMM ----------------
// C = act(A[M][K]f16 @ WT[Npad][K]f16^T + bias); K multiple of 32, M multiple of 16.
// One wave per 16x16 tile; fragments loaded straight from global (layouts per ISA 7.12.2).
__global__ void __launch_bounds__(32)
gemm_wmma_kernel(const _Float16* __restrict__ A, const _Float16* __restrict__ WT,
                 const float* __restrict__ bias, float* __restrict__ C,
                 _Float16* __restrict__ Ch,   // optional f16 output (exclusive with C)
                 int M, int K, int Ncl, int act) {
  const int lane = threadIdx.x;
  const int n0 = blockIdx.x * 16;
  const int m0 = blockIdx.y * 16;
  const int g   = lane >> 4;   // half-wave group
  const int l15 = lane & 15;

  // A fragment row for this lane: M = m0 + (lane & 15)
  const _Float16* arow = A + (size_t)(m0 + l15) * K;
  // B fragment row: N = n0 + (lane & 15), K-contiguous (transposed weights)
  const _Float16* brow = WT + (size_t)(n0 + l15) * K;

  v8f acc = {};
  for (int k0 = 0; k0 < K; k0 += 32) {
    // lanes 0-15: pos0-7 = K k0+0..7,  pos8-15 = K k0+16..23
    // lanes16-31: pos0-7 = K k0+8..15, pos8-15 = K k0+24..31
    v8h alo = *(const v8h*)(arow + k0 + g * 8);
    v8h ahi = *(const v8h*)(arow + k0 + 16 + g * 8);
    v16h af = __builtin_shufflevector(alo, ahi,
        0, 1, 2, 3, 4, 5, 6, 7, 8, 9, 10, 11, 12, 13, 14, 15);
    // B: same K order as A (pos0-7 = K k0+g*8.., pos8-15 = K k0+16+g*8..)
    v8h blo = *(const v8h*)(brow + k0 + g * 8);
    v8h bhi = *(const v8h*)(brow + k0 + 16 + g * 8);
    v16h bf = __builtin_shufflevector(blo, bhi,
        0, 1, 2, 3, 4, 5, 6, 7, 8, 9, 10, 11, 12, 13, 14, 15);
    acc = __builtin_amdgcn_wmma_f32_16x16x32_f16(
        false, af, false, bf, (short)0, acc, false, false);
  }

  // C/D layout: lane L: N = L&15; VGPR r: M = r + (L>=16)*8
  int col = n0 + l15;
  if (col < Ncl) {
    float bv = bias ? bias[col] : 0.f;
    for (int r = 0; r < 8; ++r) {
      int row = m0 + g * 8 + r;
      float v = acc[r] + bv;
      if (act == 1) v = v > 0.f ? v : 0.f;            // relu
      else if (act == 2) v = tanhf(v);                // tanh
      if (Ch) Ch[(size_t)row * Ncl + col] = (_Float16)v;
      else    C [(size_t)row * Ncl + col] = v;
    }
  }
}

// ---------------- edge kernels (GATv2 attention) ----------------
__device__ inline void atomicMaxF(float* addr, float val) {
  if (val >= 0.f) atomicMax((int*)addr, __float_as_int(val));
  else            atomicMin((unsigned int*)addr, (unsigned int)__float_as_int(val));
}

// one thread per (edge, head): score = sum_c leaky(xl[src]+xr[dst]+ea@We) * att
__global__ void edge_score_kernel(const float* __restrict__ XL,
                                  const float* __restrict__ XR,
                                  const float* __restrict__ ea,
                                  const float* __restrict__ We,
                                  const float* __restrict__ att,
                                  const int* __restrict__ ei,
                                  float* __restrict__ SB,
                                  float* __restrict__ NMAX,
                                  int H, int Cc, int HC) {
  int tid = blockIdx.x * blockDim.x + threadIdx.x;
  if (tid >= NEDGES * H) return;
  int e = tid / H, h = tid % H;
  int src = ei[e], dst = ei[NEDGES + e];
  float ek[8];
  for (int k = 0; k < 8; ++k) ek[k] = ea[(size_t)e * 8 + k];
  float s = 0.f;
  int base = h * Cc;
  for (int c = 0; c < Cc; ++c) {
    int col = base + c;
    float xev = 0.f;
    for (int k = 0; k < 8; ++k) xev += ek[k] * We[k * HC + col];
    float m = XL[(size_t)src * HC + col] + XR[(size_t)dst * HC + col] + xev;
    m = m > 0.f ? m : 0.2f * m;                 // leaky_relu(0.2)
    s += m * att[col];
  }
  SB[tid] = s;
  atomicMaxF(&NMAX[(size_t)dst * H + h], s);
}

__global__ void edge_exp_kernel(float* __restrict__ SB,
                                const float* __restrict__ NMAX,
                                float* __restrict__ NDEN,
                                const int* __restrict__ ei, int H) {
  int tid = blockIdx.x * blockDim.x + threadIdx.x;
  if (tid >= NEDGES * H) return;
  int e = tid / H, h = tid % H;
  int dst = ei[NEDGES + e];
  float ex = __expf(SB[tid] - NMAX[(size_t)dst * H + h]);
  SB[tid] = ex;
  atomicAdd(&NDEN[(size_t)dst * H + h], ex);
}

// out[dst] += xl[src] * alpha  (segment_sum)
__global__ void edge_accum_kernel(const float* __restrict__ XL,
                                  const float* __restrict__ SB,
                                  const float* __restrict__ NDEN,
                                  const int* __restrict__ ei,
                                  float* __restrict__ OUTB,
                                  int H, int Cc, int HC) {
  int tid = blockIdx.x * blockDim.x + threadIdx.x;
  if (tid >= NEDGES * H) return;
  int e = tid / H, h = tid % H;
  int src = ei[e], dst = ei[NEDGES + e];
  float alpha = SB[tid] / (NDEN[(size_t)dst * H + h] + 1e-16f);
  int base = h * Cc;
  for (int c = 0; c < Cc; ++c) {
    atomicAdd(&OUTB[(size_t)dst * HC + base + c],
              XL[(size_t)src * HC + base + c] * alpha);
  }
}

// ---------------- batchnorm ----------------
__global__ void bn_stats_kernel(const float* __restrict__ X, int HC,
                                float* __restrict__ MEAN, float* __restrict__ VAR) {
  __shared__ float ssum[256], ssq[256];
  int c = blockIdx.x, t = threadIdx.x;
  float s = 0.f, s2 = 0.f;
  for (int i = t; i < NNODES; i += 256) {
    float v = X[(size_t)i * HC + c];
    s += v; s2 += v * v;
  }
  ssum[t] = s; ssq[t] = s2;
  __syncthreads();
  for (int off = 128; off > 0; off >>= 1) {
    if (t < off) { ssum[t] += ssum[t + off]; ssq[t] += ssq[t + off]; }
    __syncthreads();
  }
  if (t == 0) {
    float mu = ssum[0] / (float)NNODES;
    MEAN[c] = mu;
    VAR[c] = ssq[0] / (float)NNODES - mu * mu;
  }
}

// normalize + ELU, emit f16 activations for the next WMMA GEMM
__global__ void bn_apply_elu_kernel(const float* __restrict__ X,
                                    const float* __restrict__ MEAN,
                                    const float* __restrict__ VAR,
                                    const float* __restrict__ gam,
                                    const float* __restrict__ bet,
                                    _Float16* __restrict__ Y, int HC) {
  int tid = blockIdx.x * blockDim.x + threadIdx.x;
  int total = NNODES * HC;
  if (tid >= total) return;
  int c = tid % HC;
  float v = (X[tid] - MEAN[c]) * rsqrtf(VAR[c] + 1e-5f) * gam[c] + bet[c];
  v = v > 0.f ? v : (__expf(v) - 1.f);   // ELU
  Y[tid] = (_Float16)v;
}

// ---------------- per-graph mean pool (f16 in, f16 out for head GEMMs) ------
__global__ void pool_kernel(const _Float16* __restrict__ X, _Float16* __restrict__ G) {
  int g = blockIdx.x, c = threadIdx.x;   // 128 threads
  float s = 0.f;
  for (int i = 0; i < NPG; ++i) s += (float)X[((size_t)g * NPG + i) * 128 + c];
  G[(size_t)g * 128 + c] = (_Float16)(s * (1.f / (float)NPG));
}

// ---------------- host orchestration ----------------
extern "C" void kernel_launch(void* const* d_in, const int* in_sizes, int n_in,
                              void* d_out, int out_size, void* d_ws, size_t ws_size,
                              hipStream_t stream) {
  if (n_in < 40) return;
  const float* x  = (const float*)d_in[0];
  const int*   ei = (const int*)d_in[1];
  const float* ea = (const float*)d_in[2];
  // d_in[3] = batch (unused: graphs are contiguous 64-node blocks)
  auto P = [&](int i) { return (const float*)d_in[4 + i]; };
  const float *bn1b = P(0),  *bn1g = P(1),  *bn2b = P(2),  *bn2g = P(3);
  const float *bn3b = P(4),  *bn3g = P(5);
  const float *c1We = P(6),  *c1Wl = P(7),  *c1Wr = P(8),  *c1att = P(9);
  const float *c2We = P(11), *c2Wl = P(12), *c2Wr = P(13), *c2att = P(14);
  const float *c3We = P(16), *c3Wl = P(17), *c3Wr = P(18), *c3att = P(19);
  const float *emb  = P(21);
  const float *pfW  = P(22), *pfb = P(23), *ppW = P(24), *ppb = P(25);
  const float *ptW  = P(26), *ptb = P(27), *strW = P(28), *strb = P(29);
  const float *tacW = P(30), *tacb = P(31), *v1W = P(32), *v1b = P(33);
  const float *v2W  = P(34), *v2b = P(35);

  // ---- workspace carve (fp32 region, then f16 region) ----
  float* ws = (float*)d_ws;
  const size_t NB = (size_t)NNODES * 256;
  float* XL   = ws;                                   // N x 256
  float* XR   = ws + NB;                              // N x 256
  float* OUTB = ws + 2 * NB;                          // N x 256
  float* SB   = ws + 3 * NB;                          // E x 4
  float* NMAX = SB + (size_t)NEDGES * 4;              // N x 4
  float* NDEN = NMAX + (size_t)NNODES * 4;            // N x 4
  float* MEAN = NDEN + (size_t)NNODES * 4;            // 256
  float* VARB = MEAN + 256;                           // 256
  _Float16* hp = (_Float16*)(VARB + 256);
  size_t hc = 0;
  auto carve = [&](size_t n) { _Float16* p = hp + hc; hc += (n + 15) & ~(size_t)15; return p; };
  _Float16* ACTH = carve((size_t)NNODES * 256);       // f16 activations (per-layer stride)
  _Float16* GPH  = carve((size_t)NGRAPH * 128);
  _Float16* V1H  = carve((size_t)NGRAPH * 128);
  _Float16* W1L = carve(128 * 32),  *W1R = carve(128 * 32);
  _Float16* W2L = carve(256 * 128), *W2R = carve(256 * 128);
  _Float16* W3L = carve(128 * 256), *W3R = carve(128 * 256);
  _Float16* WV1 = carve(128 * 128), *WV2 = carve(16 * 128);
  _Float16* WPF = carve(64 * 128),  *WPT = carve(64 * 128);
  _Float16* WPP = carve(16 * 128),  *WTC = carve(16 * 128), *WST = carve(16 * 128);
  float* out = (float*)d_out;

  auto cdiv = [](int a, int b) { return (a + b - 1) / b; };
  auto prep = [&](const float* W, _Float16* WT, int K, int N, int Kpad, int Npad) {
    prep_wt_kernel<<<cdiv(Kpad * Npad, 256), 256, 0, stream>>>(W, WT, K, N, Kpad, Npad);
  };
  auto gemm = [&](const _Float16* A, const _Float16* WT, const float* bias,
                  float* C, _Float16* Ch, int M, int K, int Ncl, int act) {
    dim3 grid(cdiv(Ncl, 16), M / 16);
    gemm_wmma_kernel<<<grid, 32, 0, stream>>>(A, WT, bias, C, Ch, M, K, Ncl, act);
  };

  // ---- weight prep (f16, transposed, padded) ----
  prep(c1Wl, W1L, 28, 128, 32, 128);   prep(c1Wr, W1R, 28, 128, 32, 128);
  prep(c2Wl, W2L, 128, 256, 128, 256); prep(c2Wr, W2R, 128, 256, 128, 256);
  prep(c3Wl, W3L, 256, 128, 256, 128); prep(c3Wr, W3R, 256, 128, 256, 128);
  prep(v1W, WV1, 128, 128, 128, 128);  prep(v2W, WV2, 128, 1, 128, 16);
  prep(pfW, WPF, 128, 64, 128, 64);    prep(ptW, WPT, 128, 64, 128, 64);
  prep(ppW, WPP, 128, 5, 128, 16);
  prep(tacW, WTC, 128, 1, 128, 16);    prep(strW, WST, 128, 1, 128, 16);

  // node features -> f16 padded h0 [N][32]
  embed_kernel<<<cdiv(NNODES, 256), 256, 0, stream>>>(x, emb, ACTH);

  struct Layer {
    const _Float16 *Wl, *Wr; const float *We, *att, *g, *b;
    int Kpad, H, C;
  } L[3] = {
    { W1L, W1R, c1We, c1att, bn1g, bn1b, 32, 4, 32 },
    { W2L, W2R, c2We, c2att, bn2g, bn2b, 128, 4, 64 },
    { W3L, W3R, c3We, c3att, bn3g, bn3b, 256, 1, 128 },
  };

  for (int l = 0; l < 3; ++l) {
    int H = L[l].H, C = L[l].C, HC = H * C, Kp = L[l].Kpad;
    int EH = NEDGES * H;
    // xl = h @ Wl ; xr = h @ Wr  (WMMA, fp32 outputs for attention math)
    gemm(ACTH, L[l].Wl, nullptr, XL, nullptr, NNODES, Kp, HC, 0);
    gemm(ACTH, L[l].Wr, nullptr, XR, nullptr, NNODES, Kp, HC, 0);
    // init segment buffers
    fill_kernel<<<cdiv(NNODES * H, 256), 256, 0, stream>>>(NMAX, -3.4e38f, NNODES * H);
    fill_kernel<<<cdiv(NNODES * H, 256), 256, 0, stream>>>(NDEN, 0.f, NNODES * H);
    fill_kernel<<<cdiv(NNODES * HC, 256), 256, 0, stream>>>(OUTB, 0.f, NNODES * HC);
    // attention scores + segment max / softmax / weighted segment sum
    edge_score_kernel<<<cdiv(EH, 256), 256, 0, stream>>>(
        XL, XR, ea, L[l].We, L[l].att, ei, SB, NMAX, H, C, HC);
    edge_exp_kernel<<<cdiv(EH, 256), 256, 0, stream>>>(SB, NMAX, NDEN, ei, H);
    edge_accum_kernel<<<cdiv(EH, 256), 256, 0, stream>>>(
        XL, SB, NDEN, ei, OUTB, H, C, HC);
    // batchnorm + ELU -> f16 activations for next GEMM
    bn_stats_kernel<<<HC, 256, 0, stream>>>(OUTB, HC, MEAN, VARB);
    bn_apply_elu_kernel<<<cdiv(NNODES * HC, 256), 256, 0, stream>>>(
        OUTB, MEAN, VARB, L[l].g, L[l].b, ACTH, HC);
  }

  // global mean pool (graphs are contiguous 64-node blocks)
  pool_kernel<<<NGRAPH, 128, 0, stream>>>(ACTH, GPH);

  // heads -> d_out at tuple offsets (value | pf | pt | pp | tac | str)
  gemm(GPH, WV1, v1b, nullptr, V1H, NGRAPH, 128, 128, 1);      // relu, f16 out
  gemm(V1H, WV2, v2b, out + 0, nullptr, NGRAPH, 128, 1, 2);    // tanh -> value [1024]
  gemm(GPH, WPF, pfb, out + 1024, nullptr, NGRAPH, 128, 64, 0);   // pf [65536]
  gemm(GPH, WPT, ptb, out + 66560, nullptr, NGRAPH, 128, 64, 0);  // pt [65536]
  gemm(GPH, WPP, ppb, out + 132096, nullptr, NGRAPH, 128, 5, 0);  // pp [5120]
  gemm(GPH, WTC, tacb, out + 137216, nullptr, NGRAPH, 128, 1, 0); // tac [1024]
  gemm(GPH, WST, strb, out + 138240, nullptr, NGRAPH, 128, 1, 0); // str [1024]
}
